// CAM_Module_8297876816357
// MI455X (gfx1250) — matrix operand order repaired and checked by reading, not compiled, not run on hardware
//
#include <hip/hip_runtime.h>

// CAM (position-attention) module, fused flash-style softmin attention for
// MI455X / gfx1250 (wave32, WMMA, 320KB LDS/WGP, async global->LDS copies).
//
//   energy = Q Q^T (Q = x[b] as [4096, 512]); attn = softmax(rowmax - energy)
//   (== softmin over energy rows); out = alpha * (attn @ Q) + x
//
// Pre-pass: x f32 -> f16 twice into d_ws (row-major + channel-major).
// Main: 4 waves/WG, 16 query rows per wave; Q frags + O accumulator register
// resident; K/V tiles double-buffered in LDS via GLOBAL_LOAD_ASYNC_TO_LDS_B128
// overlapped with WMMA; B/V fragments software-pipelined through 4 rotating
// register slots so WMMAs wait on dscnt<=6 instead of 0.

typedef _Float16 v16h __attribute__((ext_vector_type(16)));
typedef _Float16 v8h  __attribute__((ext_vector_type(8)));
typedef _Float16 v4h  __attribute__((ext_vector_type(4)));
typedef float    v8f  __attribute__((ext_vector_type(8)));
typedef int      v4i  __attribute__((ext_vector_type(4)));

#define BATCH   8
#define NPOS    4096
#define CH      512
#define ROWS_WG 64        // query rows per workgroup (16 per wave)
#define MT      64        // keys per tile
#define KS      520       // K-tile row stride in halfs (1040B = 65*16)
#define TS      72        // V^T-tile row stride in halfs (144B = 9*16)
#define PS      72        // P-stage row stride in halfs
#define NWAVES  4

static constexpr int KS_SZ    = MT * KS;          // 33280
static constexpr int VT_SZ    = CH * TS;          // 36864
static constexpr int BUF_SZ   = KS_SZ + VT_SZ;    // 70144
static constexpr int PS_OFF   = 2 * BUF_SZ;       // 140288
static constexpr int SMEM_HALFS = PS_OFF + NWAVES * 16 * PS;  // 144896
static_assert(SMEM_HALFS * 2 <= 320 * 1024, "LDS budget exceeded");

static constexpr size_t XH_ELEMS  = (size_t)BATCH * NPOS * CH;
static constexpr size_t WS_NEEDED = 2 * XH_ELEMS * sizeof(_Float16);  // 64 MiB

__device__ __forceinline__ v16h pack16(v8h lo, v8h hi) {
  union { v16h v; v8h h[2]; } u;
  u.h[0] = lo; u.h[1] = hi;
  return u.v;
}

// ---- CDNA5 async global->LDS copy (16B per lane, per-lane addresses) ------
__device__ __forceinline__ void async_copy_b128(const _Float16* __restrict__ g,
                                                _Float16* __restrict__ l) {
#if __has_builtin(__builtin_amdgcn_global_load_async_to_lds_b128)
  __builtin_amdgcn_global_load_async_to_lds_b128(
      (__attribute__((address_space(1))) v4i*)g,
      (__attribute__((address_space(3))) v4i*)l, 0, 0);
#else
  unsigned loff = (unsigned)(unsigned long long)
      (__attribute__((address_space(3))) void*)l;
  asm volatile("global_load_async_to_lds_b128 %0, %1, off"
               :: "v"(loff), "v"(g) : "memory");
#endif
}

__device__ __forceinline__ void wait_async0() {
  asm volatile("s_wait_asynccnt 0x0" ::: "memory");
}

// ============================ pre-pass kernel ===============================
__global__ __launch_bounds__(256)
void cam_cvt_transpose_kernel(const float* __restrict__ x,
                              _Float16* __restrict__ xh,
                              _Float16* __restrict__ xt) {
  __shared__ _Float16 tile[64][72];
  const int bid = blockIdx.x;
  const int cb  = bid & 7;            // 512/64 channel tiles
  const int nb  = (bid >> 3) & 63;    // 4096/64 row tiles
  const int b   = bid >> 9;

  const float* src = x + ((size_t)b * NPOS + nb * 64) * CH + cb * 64;

  for (int j = threadIdx.x; j < 64 * 16; j += 256) {
    const int r = j >> 4, c4 = (j & 15) * 4;
    const float4 v = *(const float4*)(src + (size_t)r * CH + c4);
    v4h h;
    h[0] = (_Float16)v.x; h[1] = (_Float16)v.y;
    h[2] = (_Float16)v.z; h[3] = (_Float16)v.w;
    *(v4h*)(xh + ((size_t)b * NPOS + nb * 64 + r) * CH + cb * 64 + c4) = h;
    tile[r][c4 + 0] = h[0]; tile[r][c4 + 1] = h[1];
    tile[r][c4 + 2] = h[2]; tile[r][c4 + 3] = h[3];
  }
  __syncthreads();
  for (int j = threadIdx.x; j < 64 * 16; j += 256) {
    const int c = j >> 4, n4 = (j & 15) * 4;
    v4h h;
    h[0] = tile[n4 + 0][c]; h[1] = tile[n4 + 1][c];
    h[2] = tile[n4 + 2][c]; h[3] = tile[n4 + 3][c];
    *(v4h*)(xt + ((size_t)b * CH + cb * 64 + c) * NPOS + nb * 64 + n4) = h;
  }
}

// ============================== main kernel =================================
__device__ __forceinline__ void fill_tile_async(const _Float16* __restrict__ xh_b,
                                                const _Float16* __restrict__ xt_b,
                                                _Float16* __restrict__ ksbuf,
                                                _Float16* __restrict__ vtbuf,
                                                int m0, int tid) {
  for (int j = tid; j < MT * (CH / 8); j += 128) {          // K rows
    const int r = j >> 6, c8 = (j & 63) * 8;
    async_copy_b128(xh_b + (size_t)(m0 + r) * CH + c8, ksbuf + r * KS + c8);
  }
  for (int j = tid; j < CH * (MT / 8); j += 128) {          // V transposed
    const int c = j >> 3, k8 = (j & 7) * 8;
    async_copy_b128(xt_b + (size_t)c * NPOS + m0 + k8, vtbuf + c * TS + k8);
  }
}

__global__ __launch_bounds__(128, 1)
void cam_flash_softmin_kernel(const float* __restrict__ x,
                              const _Float16* __restrict__ xh,
                              const _Float16* __restrict__ xt,
                              const float* __restrict__ alphaP,
                              float* __restrict__ out) {
  extern __shared__ _Float16 sm[];

  const int tid  = threadIdx.x;
  const int lane = tid & 31;
  const int wave = tid >> 5;
  const int half = (lane >> 4) & 1;
  const int l16  = lane & 15;

  const int b    = blockIdx.x / (NPOS / ROWS_WG);
  const int row0 = (blockIdx.x % (NPOS / ROWS_WG)) * ROWS_WG;

  const float      alpha = alphaP[0];
  const _Float16*  xh_b  = xh + (size_t)b * NPOS * CH;
  const _Float16*  xt_b  = xt + (size_t)b * CH * NPOS;

  _Float16* ps = sm + PS_OFF + wave * 16 * PS;

  // ---- Q A-fragments (16 rows x 512 ch) from the f16 global copy
  v16h qa[16];
  {
    const _Float16* qrow = xh_b + (size_t)(row0 + wave * 16 + l16) * CH;
#pragma unroll
    for (int kc = 0; kc < 16; ++kc) {
      const v8h lo = *(const v8h*)(qrow + kc * 32 + (half ? 16 : 0));
      const v8h hi = *(const v8h*)(qrow + kc * 32 + (half ? 24 : 8));
      qa[kc] = pack16(lo, hi);
    }
  }

  v8f o[32];
#pragma unroll
  for (int i = 0; i < 32; ++i) o[i] = (v8f){0, 0, 0, 0, 0, 0, 0, 0};
  v8f rm, lsum;
#pragma unroll
  for (int e = 0; e < 8; ++e) { rm[e] = __builtin_inff(); lsum[e] = 0.0f; }

  fill_tile_async(xh_b, xt_b, sm, sm + KS_SZ, 0, tid);

  for (int mi = 0; mi < NPOS / MT; ++mi) {
    _Float16* ksbuf = sm + (mi & 1) * BUF_SZ;
    _Float16* vtbuf = ksbuf + KS_SZ;

    wait_async0();
    __syncthreads();

    if (mi + 1 < NPOS / MT) {
      _Float16* nks = sm + ((mi + 1) & 1) * BUF_SZ;
      fill_tile_async(xh_b, xt_b, nks, nks + KS_SZ, (mi + 1) * MT, tid);
    }

    // ---- S = Q K^T : software-pipelined, 8 independent accumulator chains
    const _Float16* kbase = ksbuf + l16 * KS + (half ? 16 : 0);
    auto loadBfrag = [&](int u) -> v16h {
      const int kc = u >> 2, t = u & 3;
      const _Float16* krow = kbase + t * 16 * KS + kc * 32;
      return pack16(*(const v8h*)(krow), *(const v8h*)(krow + 8));
    };

    v8f s[4], s2[4];
#pragma unroll
    for (int t = 0; t < 4; ++t) {
      s[t]  = (v8f){0, 0, 0, 0, 0, 0, 0, 0};
      s2[t] = (v8f){0, 0, 0, 0, 0, 0, 0, 0};
    }

    v16h bst[4];
#pragma unroll
    for (int u = 0; u < 4; ++u) bst[u] = loadBfrag(u);
#pragma unroll
    for (int u = 0; u < 64; ++u) {
      const int kc = u >> 2, t = u & 3;
      const v16h bcur = bst[u & 3];
      if (u + 4 < 64) bst[u & 3] = loadBfrag(u + 4);   // prefetch 4 ahead
      if (u & 4)
        s2[t] = __builtin_amdgcn_wmma_f32_16x16x32_f16(
            false, qa[kc], false, bcur, (short)0, s2[t], false, false);
      else
        s[t]  = __builtin_amdgcn_wmma_f32_16x16x32_f16(
            false, qa[kc], false, bcur, (short)0, s[t],  false, false);
    }
#pragma unroll
    for (int t = 0; t < 4; ++t)
#pragma unroll
      for (int e = 0; e < 8; ++e) s[t][e] += s2[t][e];

    // ---- online softmin stats (C-layout: element r <-> row r + 8*half)
    v8f tm;
#pragma unroll
    for (int e = 0; e < 8; ++e)
      tm[e] = __builtin_fminf(__builtin_fminf(s[0][e], s[1][e]),
                              __builtin_fminf(s[2][e], s[3][e]));
#pragma unroll
    for (int msk = 1; msk <= 8; msk <<= 1)
#pragma unroll
      for (int e = 0; e < 8; ++e)
        tm[e] = __builtin_fminf(tm[e], __shfl_xor(tm[e], msk, 32));

    v8f nm, corr;
#pragma unroll
    for (int e = 0; e < 8; ++e) {
      nm[e]   = __builtin_fminf(rm[e], tm[e]);
      corr[e] = __expf(nm[e] - rm[e]);   // exp(-inf)=0 handles the first tile
      rm[e]   = nm[e];
    }

    // ---- P = exp(min - S); row-sum; stage P to LDS first (stores overlap
    //      with the O-rescale VALU work below)
#pragma unroll
    for (int t = 0; t < 4; ++t)
#pragma unroll
      for (int e = 0; e < 8; ++e) s[t][e] = __expf(nm[e] - s[t][e]);

#pragma unroll
    for (int t = 0; t < 4; ++t)
#pragma unroll
      for (int r = 0; r < 8; ++r)
        ps[(r + 8 * half) * PS + t * 16 + l16] = (_Float16)s[t][r];

    v8f rs;
#pragma unroll
    for (int e = 0; e < 8; ++e) rs[e] = s[0][e] + s[1][e] + s[2][e] + s[3][e];
#pragma unroll
    for (int msk = 1; msk <= 8; msk <<= 1)
#pragma unroll
      for (int e = 0; e < 8; ++e) rs[e] += __shfl_xor(rs[e], msk, 32);
#pragma unroll
    for (int e = 0; e < 8; ++e) lsum[e] = lsum[e] * corr[e] + rs[e];

    // ---- rescale O only when the running min actually changed (corr != 1).
    // After the first tiles this is rare; v_exp(0)==1.0 exactly so the
    // wave-uniform ballot test is sound.
    float cmin = 1.0f;
#pragma unroll
    for (int e = 0; e < 8; ++e) cmin = __builtin_fminf(cmin, corr[e]);
    if (__ballot(cmin < 1.0f)) {
#pragma unroll
      for (int i = 0; i < 32; ++i)
#pragma unroll
        for (int e = 0; e < 8; ++e) o[i][e] *= corr[e];
    }

    // ---- O += P V : software-pipelined V fragments, 32 accumulator chains
    v16h pa[2];
#pragma unroll
    for (int kt = 0; kt < 2; ++kt) {
      const _Float16* prow = ps + l16 * PS + kt * 32;
      pa[kt] = pack16(*(const v8h*)(prow + (half ? 16 : 0)),
                      *(const v8h*)(prow + (half ? 24 : 8)));
    }
    auto loadVfrag = [&](int u) -> v16h {
      const int kt = u >> 5, nt = u & 31;
      const _Float16* vrow = vtbuf + (nt * 16 + l16) * TS + kt * 32 + (half ? 16 : 0);
      return pack16(*(const v8h*)(vrow), *(const v8h*)(vrow + 8));
    };
    v16h vst[4];
#pragma unroll
    for (int u = 0; u < 4; ++u) vst[u] = loadVfrag(u);
#pragma unroll
    for (int u = 0; u < 64; ++u) {
      const int nt = u & 31;
      const v16h vcur = vst[u & 3];
      if (u + 4 < 64) vst[u & 3] = loadVfrag(u + 4);   // prefetch 4 ahead
      o[nt] = __builtin_amdgcn_wmma_f32_16x16x32_f16(
          false, pa[u >> 5], false, vcur, (short)0, o[nt], false, false);
    }

    __syncthreads();   // everyone done with this buffer before it is refilled
  }

  // ---- epilogue: y = alpha * (O / l) + x
  const int rbase = row0 + wave * 16 + 8 * half;
#pragma unroll
  for (int nt = 0; nt < 32; ++nt) {
    const int chn = nt * 16 + l16;
#pragma unroll
    for (int r = 0; r < 8; ++r) {
      const size_t idx = ((size_t)b * NPOS + (rbase + r)) * CH + chn;
      out[idx] = alpha * (o[nt][r] / lsum[r]) + x[idx];
    }
  }
}

extern "C" void kernel_launch(void* const* d_in, const int* in_sizes, int n_in,
                              void* d_out, int out_size, void* d_ws, size_t ws_size,
                              hipStream_t stream) {
  (void)in_sizes; (void)n_in; (void)out_size;
  const float* x     = (const float*)d_in[0];
  const float* alpha = (const float*)d_in[1];
  float*       out   = (float*)d_out;

  _Float16* xh = (_Float16*)d_ws;              // f16 row-major copy
  _Float16* xt = xh + XH_ELEMS;                // f16 channel-major copy
  (void)ws_size;                               // needs WS_NEEDED = 64 MiB

  cam_cvt_transpose_kernel<<<dim3(BATCH * 64 * 8), dim3(256), 0, stream>>>(x, xh, xt);

  const size_t smem = (size_t)SMEM_HALFS * sizeof(_Float16);  // ~283 KB
  cam_flash_softmin_kernel<<<dim3(BATCH * (NPOS / ROWS_WG)), dim3(128), smem, stream>>>(
      x, xh, xt, alpha, out);
}
